// SA_39213051412691
// MI455X (gfx1250) — matrix operand order repaired and checked by reading, not compiled, not run on hardware
//
#include <hip/hip_runtime.h>
#include <hip/hip_bf16.h>

// ---------------------------------------------------------------------------
// MI455X (gfx1250) spatial-reduction attention, wave32 + WMMA f16->f32.
//
// B=4, N=9216 (96x96), C=hd=64, SR=2 -> N'=2304, heads=1, scale folded into q.
// Attention (~22 GFLOP) is compute-bound (HBM traffic ~20MB ≈ 1us @23.3TB/s),
// so both GEMMs of the attention run on v_wmma_f32_16x16x32_f16 with f32
// accumulators, flash-style (S/P never leave the WGP). K/V tiles stream into
// LDS via the CDNA5 async-to-LDS DMA path (ASYNCcnt) with double buffering.
// ---------------------------------------------------------------------------

typedef __attribute__((ext_vector_type(16))) _Float16 v16h;
typedef __attribute__((ext_vector_type(8)))  float    v8f;
typedef __attribute__((ext_vector_type(4)))  int      v4i;

#define WMMA_F16(A, B, C) \
  __builtin_amdgcn_wmma_f32_16x16x32_f16(false, (A), false, (B), (short)0, (C), false, false)

static constexpr int   BATCH = 4;
static constexpr int   NQ    = 9216;   // 96*96 queries
static constexpr int   NK    = 2304;   // 48*48 keys
static constexpr int   CH    = 64;
static constexpr float SCALE = 0.125f; // 64^-0.5, folded into q projection
static constexpr float EPSV  = 1e-5f;

// ----- CDNA5 async global->LDS staging (ASYNCcnt), with safe fallback ------
#if __has_builtin(__builtin_amdgcn_global_load_async_to_lds_b128)
#define ASYNC_LDS 1
#endif

#ifdef ASYNC_LDS
#if __has_builtin(__builtin_amdgcn_s_wait_asynccnt)
#define WAIT_ASYNC(n) __builtin_amdgcn_s_wait_asynccnt(n)
#else
#define WAIT_ASYNC(n) asm volatile("s_wait_asynccnt %0" ::"i"(n) : "memory")
#endif
#else
#define WAIT_ASYNC(n)
#endif

// Stage one 32x64 f16 tile (4 KB): 8 x b128 per lane.
__device__ __forceinline__ void stage_tile(const _Float16* __restrict__ g,
                                           _Float16* l, int lane) {
#ifdef ASYNC_LDS
  typedef __attribute__((address_space(1))) v4i GVec;   // global int4
  typedef __attribute__((address_space(3))) v4i LVec;   // LDS int4
  const v4i* gs = (const v4i*)g;
  v4i*       ld = (v4i*)l;
#pragma unroll
  for (int i = 0; i < 8; ++i)
    __builtin_amdgcn_global_load_async_to_lds_b128(
        (GVec*)(gs + lane + 32 * i), (LVec*)(ld + lane + 32 * i), 0, 0);
#else
  const uint4* gs = (const uint4*)g;
  uint4*       ld = (uint4*)l;
#pragma unroll
  for (int i = 0; i < 8; ++i) ld[lane + 32 * i] = gs[lane + 32 * i];
#endif
}

// ---------------------------------------------------------------------------
// Kernel 1: q = (x @ wq^T) * SCALE  (f32 in, f16 out). One wave per 16-row
// strip; loops all 4 column tiles so A operands/loads are reused 4x.
// A layout (16-bit 16x32): lane r=lane&15 holds row r; elem e -> K =
//   16*(lane>=16) + e.  B layout (32x16): lane holds col r,
//   elem e -> K = 16*(lane>=16) + e.  (contiguous k order, used for both A and B)
// ---------------------------------------------------------------------------
__global__ __launch_bounds__(32) void qproj_kernel(
    const float* __restrict__ x, const float* __restrict__ wq,
    _Float16* __restrict__ q) {
  const int lane = threadIdx.x, hi = lane >> 4, r = lane & 15;
  const size_t row0 = (size_t)blockIdx.x * 16;

  const float* xr = x + (row0 + r) * CH;
  v16h a0, a1;
#pragma unroll
  for (int e = 0; e < 16; ++e) {
    const int Ka = hi * 16 + e;
    a0[e] = (_Float16)xr[Ka];
    a1[e] = (_Float16)xr[32 + Ka];
  }
#pragma unroll
  for (int ct = 0; ct < 4; ++ct) {
    const float* wr = wq + (size_t)(ct * 16 + r) * CH;  // B = wq^T -> wq rows
    v16h b0, b1;
#pragma unroll
    for (int e = 0; e < 16; ++e) {
      const int Kb = hi * 16 + e;
      b0[e] = (_Float16)wr[Kb];
      b1[e] = (_Float16)wr[32 + Kb];
    }
    v8f c = {};
    c = WMMA_F16(a0, b0, c);
    c = WMMA_F16(a1, b1, c);
#pragma unroll
    for (int v = 0; v < 8; ++v)                         // C/D: row=v+8*hi
      q[(row0 + v + 8 * hi) * CH + ct * 16 + r] = (_Float16)(c[v] * SCALE);
  }
}

// ---------------------------------------------------------------------------
// Kernel 2: spatial reduce: conv2x2/s2 + BN + LN + kv projection.
// One 128-thread block per reduced pixel (B*48*48 = 9216 blocks, ~1 GF total).
// ---------------------------------------------------------------------------
__global__ __launch_bounds__(128) void kv_reduce_kernel(
    const float* __restrict__ x,  const float* __restrict__ cw,
    const float* __restrict__ cb, const float* __restrict__ bng,
    const float* __restrict__ bnb,const float* __restrict__ bnm,
    const float* __restrict__ bnv,const float* __restrict__ lng,
    const float* __restrict__ lnb,const float* __restrict__ wkv,
    _Float16* __restrict__ kf, _Float16* __restrict__ vf) {
  __shared__ float xwin[256];      // 2x2 window, [pix][ch]
  __shared__ float xn[64];         // normalized channels
  __shared__ float s1_[64], s2_[64];

  const int t   = threadIdx.x;
  const int bid = blockIdx.x;
  const int b   = bid / NK;
  const int rem = bid % NK;        // = i*48 + j
  const int i   = rem / 48, j = rem % 48;
  const float* xb = x + (size_t)b * NQ * CH;

  for (int idx = t; idx < 256; idx += 128) {
    const int p = idx >> 6, c = idx & 63;               // p = di*2+dj
    const int n = (2 * i + (p >> 1)) * 96 + 2 * j + (p & 1);
    xwin[idx] = xb[(size_t)n * CH + c];
  }
  __syncthreads();

  if (t < 64) {
    float acc = cb[t];
    const float* w = cw + (size_t)t * 256;              // [c_in*4 + di*2 + dj]
#pragma unroll 4
    for (int ci = 0; ci < 64; ++ci) {
      acc += xwin[ci]       * w[ci * 4 + 0];
      acc += xwin[64 + ci]  * w[ci * 4 + 1];
      acc += xwin[128 + ci] * w[ci * 4 + 2];
      acc += xwin[192 + ci] * w[ci * 4 + 3];
    }
    const float y = (acc - bnm[t]) * rsqrtf(bnv[t] + EPSV) * bng[t] + bnb[t];
    xn[t] = y;  s1_[t] = y;  s2_[t] = y * y;
  }
  __syncthreads();
  for (int st = 32; st > 0; st >>= 1) {
    if (t < st) { s1_[t] += s1_[t + st]; s2_[t] += s2_[t + st]; }
    __syncthreads();
  }
  const float mu   = s1_[0] * (1.f / 64.f);
  const float var  = s2_[0] * (1.f / 64.f) - mu * mu;
  const float rsig = rsqrtf(var + EPSV);
  __syncthreads();
  if (t < 64) xn[t] = (xn[t] - mu) * rsig * lng[t] + lnb[t];
  __syncthreads();

  const float* wrow = wkv + (size_t)t * CH;             // wkv: [128,64]
  float d = 0.f;
#pragma unroll 8
  for (int ci = 0; ci < 64; ++ci) d += xn[ci] * wrow[ci];

  const size_t orow = ((size_t)b * NK + rem) * CH;
  if (t < 64) kf[orow + t]        = (_Float16)d;
  else        vf[orow + (t - 64)] = (_Float16)d;
}

// ---------------------------------------------------------------------------
// Kernel 3: flash attention. One wave per 16-query tile, 32 keys per step.
// Next K/V tiles stream via async-to-LDS DMA (double buffered) while the
// current step runs: 4 WMMA (QK^T, K-dim=channels) + online softmax
// (shfl_xor within 16-lane halves == C-layout rows) + LDS bounce to A layout
// + 4 WMMA (P.V, K-dim=keys). Async loads retire in order, so
// s_wait_asynccnt 16 == "current tile landed, next still in flight".
// ---------------------------------------------------------------------------
__global__ __launch_bounds__(32) void flash_attn_kernel(
    const _Float16* __restrict__ qf, const _Float16* __restrict__ kf,
    const _Float16* __restrict__ vf, _Float16* __restrict__ ctx) {
  __shared__ __align__(16) _Float16 kt[2][32 * 64];   // K tiles (keys x ch)
  __shared__ __align__(16) _Float16 vt[2][32 * 64];   // V tiles (keys x ch)
  __shared__ __align__(16) float    pt[16 * 32];      // P bounce (q x keys)

  const int lane = threadIdx.x, hi = lane >> 4, r = lane & 15;
  const int b = blockIdx.y;
  const size_t qrow = (size_t)b * NQ + (size_t)blockIdx.x * 16;

  // Q tile in A layout, two K=32 chunks over the 64 channels (pre-scaled).
  v16h qa0, qa1;
  {
    const _Float16* rp = qf + (qrow + r) * CH;
#pragma unroll
    for (int e = 0; e < 16; ++e) {
      const int K = hi * 16 + e;
      qa0[e] = rp[K];  qa1[e] = rp[32 + K];
    }
  }

  v8f acc[4] = {{}, {}, {}, {}};
  float m[8], l[8];
#pragma unroll
  for (int v = 0; v < 8; ++v) { m[v] = -1e30f; l[v] = 0.f; }

  const _Float16* kb_ = kf + (size_t)b * NK * CH;
  const _Float16* vb_ = vf + (size_t)b * NK * CH;

  // Prologue: stage tile 0 into buffer 0.
  stage_tile(kb_, kt[0], lane);
  stage_tile(vb_, vt[0], lane);
  int cur = 0;

  for (int j0 = 0; j0 < NK; j0 += 32) {
    const int nxt = cur ^ 1;
    const bool has_next = (j0 + 32) < NK;
    if (has_next) {                       // overlap DMA with this step's math
      stage_tile(kb_ + (size_t)(j0 + 32) * CH, kt[nxt], lane);
      stage_tile(vb_ + (size_t)(j0 + 32) * CH, vt[nxt], lane);
      WAIT_ASYNC(16);                     // current 16 b128 DMAs retired
    } else {
      WAIT_ASYNC(0);
    }
    __syncthreads();

    const _Float16* ktc = kt[cur];
    const _Float16* vtc = vt[cur];

    // S = Q K^T for two 16-key subtiles (K-dim = 64 channels, 2 steps each).
    v8f s0 = {}, s1 = {};
    {
      const _Float16* k0 = ktc + r * CH + hi * 16;         // subtile 0, col=r
      const _Float16* k1 = ktc + (16 + r) * CH + hi * 16;  // subtile 1
      v16h b00, b01, b10, b11;
#pragma unroll
      for (int e = 0; e < 16; ++e) {
        b00[e] = k0[e];  b01[e] = k0[32 + e];
        b10[e] = k1[e];  b11[e] = k1[32 + e];
      }
      s0 = WMMA_F16(qa0, b00, s0);  s0 = WMMA_F16(qa1, b01, s0);
      s1 = WMMA_F16(qa0, b10, s1);  s1 = WMMA_F16(qa1, b11, s1);
    }

    // Online softmax; each lane owns rows v+8*hi, cols live across 16 lanes.
#pragma unroll
    for (int v = 0; v < 8; ++v) {
      const float a0 = s0[v], a1 = s1[v];   // scale pre-folded into q
      float mx = fmaxf(a0, a1);
      mx = fmaxf(mx, __shfl_xor(mx, 1));
      mx = fmaxf(mx, __shfl_xor(mx, 2));
      mx = fmaxf(mx, __shfl_xor(mx, 4));
      mx = fmaxf(mx, __shfl_xor(mx, 8));
      const float mn   = fmaxf(m[v], mx);
      const float corr = __expf(m[v] - mn);
      m[v] = mn;
      const float e0 = __expf(a0 - mn), e1 = __expf(a1 - mn);
      float rs = e0 + e1;
      rs += __shfl_xor(rs, 1);
      rs += __shfl_xor(rs, 2);
      rs += __shfl_xor(rs, 4);
      rs += __shfl_xor(rs, 8);
      l[v] = l[v] * corr + rs;
      acc[0][v] *= corr; acc[1][v] *= corr; acc[2][v] *= corr; acc[3][v] *= corr;
      pt[(v + 8 * hi) * 32 + r]      = e0;  // C-layout -> row-major bounce
      pt[(v + 8 * hi) * 32 + 16 + r] = e1;
    }
    __syncthreads();

    // P in A layout (K-dim = 32 keys).
    v16h pa;
#pragma unroll
    for (int e = 0; e < 16; ++e) {
      const int K = hi * 16 + e;
      pa[e] = (_Float16)pt[r * 32 + K];
    }
    // acc += P V  (V in direct B layout: elem e -> key = 16*hi + e).
#pragma unroll
    for (int n = 0; n < 4; ++n) {
      v16h vbm;
#pragma unroll
      for (int e = 0; e < 16; ++e)
        vbm[e] = vtc[(hi * 16 + e) * CH + 16 * n + r];
      acc[n] = WMMA_F16(pa, vbm, acc[n]);
    }
    __syncthreads();   // all LDS reads of buf `cur` done before it is re-staged
    cur = nxt;
  }

  // Normalize and emit context (f16, row-major).
#pragma unroll
  for (int v = 0; v < 8; ++v) {
    const float inv = 1.f / l[v];
    const size_t grow = (qrow + v + 8 * hi) * CH;
#pragma unroll
    for (int n = 0; n < 4; ++n)
      ctx[grow + 16 * n + r] = (_Float16)(acc[n][v] * inv);
  }
}

// ---------------------------------------------------------------------------
// Kernel 4: out = ctx @ wproj^T + bproj  (f16 A, f32 out). One wave per
// 16-row strip, all 4 column tiles (A reused 4x).
// ---------------------------------------------------------------------------
__global__ __launch_bounds__(32) void proj_kernel(
    const _Float16* __restrict__ ctx, const float* __restrict__ wp,
    const float* __restrict__ bp, float* __restrict__ out) {
  const int lane = threadIdx.x, hi = lane >> 4, r = lane & 15;
  const size_t row0 = (size_t)blockIdx.x * 16;

  const _Float16* cr = ctx + (row0 + r) * CH;
  v16h a0, a1;
#pragma unroll
  for (int e = 0; e < 16; ++e) {
    const int Ka = hi * 16 + e;
    a0[e] = cr[Ka];
    a1[e] = cr[32 + Ka];
  }
#pragma unroll
  for (int ct = 0; ct < 4; ++ct) {
    const float* wr = wp + (size_t)(ct * 16 + r) * CH;
    v16h b0, b1;
#pragma unroll
    for (int e = 0; e < 16; ++e) {
      const int Kb = hi * 16 + e;
      b0[e] = (_Float16)wr[Kb];
      b1[e] = (_Float16)wr[32 + Kb];
    }
    v8f c = {};
    c = WMMA_F16(a0, b0, c);
    c = WMMA_F16(a1, b1, c);
    const float bias = bp[ct * 16 + r];
#pragma unroll
    for (int v = 0; v < 8; ++v)
      out[(row0 + v + 8 * hi) * CH + ct * 16 + r] = c[v] + bias;
  }
}

// ---------------------------------------------------------------------------
extern "C" void kernel_launch(void* const* d_in, const int* in_sizes, int n_in,
                              void* d_out, int out_size, void* d_ws, size_t ws_size,
                              hipStream_t stream) {
  const float* x      = (const float*)d_in[0];
  const float* conv_w = (const float*)d_in[1];
  const float* conv_b = (const float*)d_in[2];
  const float* bng    = (const float*)d_in[3];
  const float* bnb    = (const float*)d_in[4];
  const float* bnm    = (const float*)d_in[5];
  const float* bnv    = (const float*)d_in[6];
  const float* lng    = (const float*)d_in[7];
  const float* lnb    = (const float*)d_in[8];
  const float* wq     = (const float*)d_in[9];
  const float* wkv    = (const float*)d_in[10];
  const float* wproj  = (const float*)d_in[11];
  const float* bproj  = (const float*)d_in[12];

  const size_t MQ = (size_t)BATCH * NQ;               // 36864 rows
  const size_t MK = (size_t)BATCH * NK;               // 9216 reduced rows

  _Float16* qf  = (_Float16*)d_ws;                    // [MQ,64] f16 (pre-scaled)
  _Float16* kf  = qf + MQ * CH;                       // [MK,64] f16
  _Float16* vf  = kf + MK * CH;                       // [MK,64] f16
  _Float16* ctx = vf + MK * CH;                       // [MQ,64] f16

  qproj_kernel<<<(unsigned)(MQ / 16), 32, 0, stream>>>(x, wq, qf);
  kv_reduce_kernel<<<(unsigned)MK, 128, 0, stream>>>(
      x, conv_w, conv_b, bng, bnb, bnm, bnv, lng, lnb, wkv, kf, vf);
  flash_attn_kernel<<<dim3(NQ / 16, BATCH), 32, 0, stream>>>(qf, kf, vf, ctx);
  proj_kernel<<<(unsigned)(MQ / 16), 32, 0, stream>>>(ctx, wproj, bproj,
                                                      (float*)d_out);
}
